// DLRM_21217138442338
// MI455X (gfx1250) — matrix operand order repaired and checked by reading, not compiled, not run on hardware
//
#include <hip/hip_runtime.h>
#include <hip/hip_bf16.h>

// ---------------------------------------------------------------------------
// Types for WMMA fragments (CDNA5 wave32, V_WMMA_F32_16X16X32_F16)
// ---------------------------------------------------------------------------
typedef __attribute__((ext_vector_type(16))) _Float16 v16h;
typedef __attribute__((ext_vector_type(8)))  _Float16 v8h;
typedef __attribute__((ext_vector_type(8)))  float    v8f;

#define WAVE_LDS_WAIT() asm volatile("s_wait_dscnt 0x0" ::: "memory")

static __device__ inline v8f zero8() {
  v8f z;
#pragma unroll
  for (int i = 0; i < 8; ++i) z[i] = 0.f;
  return z;
}

static __device__ inline v8f wmma32(v16h a, v16h b, v8f c) {
  // D = A(16x32 f16) * B(32x16 f16) + C(16x16 f32)
  return __builtin_amdgcn_wmma_f32_16x16x32_f16(
      /*neg_a=*/false, a, /*neg_b=*/false, b,
      /*c_mod=*/(short)0, c, /*reuse_a=*/false, /*reuse_b=*/false);
}

// Fast branch-free transcendentals: v_exp_f32 + v_rcp_f32, no IEEE div.
static __device__ inline float fast_sigm(float x) {
  return __builtin_amdgcn_rcpf(1.f + __expf(-x));
}
static __device__ inline float fast_tanh(float x) {
  float e = __expf(-2.f * x);                     // e = exp(-2x)
  return (1.f - e) * __builtin_amdgcn_rcpf(1.f + e);
}

// Load an A fragment (16x32 f16) from row-major LDS [16][stride] halves.
// ISA layout: lane L holds row m=L%16; elements e0..7 -> K=koff+0..7,
// e8..15 -> K=koff+16..23, koff = (L<16)?0:8.
static __device__ inline v16h loadA(const _Float16* base, int stride, int kbase, int lane) {
  int m    = lane & 15;
  int koff = (lane >> 4) << 3;
  const _Float16* p = base + m * stride + kbase + koff;
  v8h lo = *(const v8h*)(p);
  v8h hi = *(const v8h*)(p + 16);
  v16h a;
#pragma unroll
  for (int e = 0; e < 8; ++e) { a[e] = lo[e]; a[e + 8] = hi[e]; }
  return a;
}

// ---------------------------------------------------------------------------
// Parameter bundle (built host-side, passed by value)
// ---------------------------------------------------------------------------
struct Params {
  const float* launch_seq;
  const float* play_seq;
  const float* dur;          // (B,16)
  const float* inter;        // (B,11)
  const float* dense;        // (B,11)
  const int*   sparse;       // (B,5)
  const float* emb[5];       // tiny tables, 16-wide
  const float *b_bot0, *b_bot1, *b_bot2, *b_bot3;
  const float *b_top0, *b_top1, *b_top2;
  const float *gl_kern, *gl_bias, *gp_kern, *gp_bias;
  const v16h *Wb0, *Wb1, *Wb2, *Wb3;          // swizzled f16 weights
  const v16h *Wt0, *Wt1, *Wt2;
  const v16h *Wgl, *Wgp;
  const float *sc_b0, *sh_b0, *sc_b1, *sh_b1, *sc_b2, *sh_b2;
  const float *sc_t0, *sh_t0, *sc_t1, *sh_t1;
  float* out;
};

// ---------------------------------------------------------------------------
// Prep kernel: row-major f32 W[K][N] -> swizzled f16 B-fragment tiles.
// Tile t = nt*KT + kt, 512 halves per tile:  dst[(t*32+lane)*16+e] =
//   W[kt*32 + (lane<16?0:8) + (e&7) + (e>=8?16:0)][nt*16 + lane%16]  (zero-padded).
// ---------------------------------------------------------------------------
__global__ void swizzle_w(const float* __restrict__ W, _Float16* __restrict__ dst,
                          int Ksrc, int Nsrc, int KT, int NT) {
  int total = KT * NT * 512;
  for (int idx = blockIdx.x * blockDim.x + threadIdx.x; idx < total;
       idx += gridDim.x * blockDim.x) {
    int t = idx >> 9, r = idx & 511;
    int lane = r >> 4, e = r & 15;
    int nt = t / KT, kt = t - nt * KT;
    int n = nt * 16 + (lane & 15);
    int k = kt * 32 + ((lane >> 4) << 3) + (e & 7) + ((e >> 3) << 4);
    float v = (k < Ksrc && n < Nsrc) ? W[k * Nsrc + n] : 0.f;
    dst[idx] = (_Float16)v;
  }
}

__global__ void bn_prep(const float* __restrict__ g, const float* __restrict__ beta,
                        const float* __restrict__ m, const float* __restrict__ v,
                        float* __restrict__ scale, float* __restrict__ shift, int n) {
  int i = blockIdx.x * blockDim.x + threadIdx.x;
  if (i < n) {
    float s = g[i] * rsqrtf(v[i] + 1e-3f);
    scale[i] = s;
    shift[i] = beta[i] - m[i] * s;
  }
}

// ---------------------------------------------------------------------------
// Per-wave LDS: 33792 bytes
// ---------------------------------------------------------------------------
struct WaveLDS {
  _Float16 bufA[16 * 512];   // 16 KB  ping
  _Float16 bufB[16 * 256];   // 8 KB   pong (aliased as float T[16*96] for interactions)
  _Float16 R[16 * 128];      // 4 KB   top-MLP input (padded to 128)
  float    seq[16 * 32];     // 2 KB   GRU input tile
  _Float16 hbf[16 * 32];     // 1 KB   GRU hidden (f16, A-fragment source); also dense staging
  float    hf32[16 * 32];    // 2 KB   GRU hidden (f32, for per-step means)
};

// out[16][Nout] = act( in[16][Kdim] @ W ) with fused bias / ReLU / BN
static __device__ void gemm16(const _Float16* __restrict__ inL, int inStride, int Kdim,
                              _Float16* __restrict__ outL, int outStride, int Nout,
                              const v16h* __restrict__ Wsw,
                              const float* __restrict__ bias,
                              const float* __restrict__ scale,
                              const float* __restrict__ shift,
                              bool relu, int lane) {
  int KT = Kdim >> 5, NT = Nout >> 4;
  int m0 = (lane >> 4) << 3;
  int nlo = lane & 15;
  for (int nt = 0; nt < NT; ++nt) {
    v8f acc = zero8();
    const v16h* wt = Wsw + (size_t)nt * KT * 32 + lane;
#pragma unroll 4
    for (int kt = 0; kt < KT; ++kt) {
      v16h a = loadA(inL, inStride, kt << 5, lane);
      v16h b = wt[kt * 32];
      acc = wmma32(a, b, acc);
    }
    int n = nt * 16 + nlo;
    float bs = bias[n];
    float sc = scale ? scale[n] : 1.f;
    float sh = shift ? shift[n] : 0.f;
#pragma unroll
    for (int r = 0; r < 8; ++r) {
      float v = acc[r] + bs;
      if (relu) v = fmaxf(v, 0.f);
      v = v * sc + sh;
      outL[(r + m0) * outStride + n] = (_Float16)v;
    }
  }
}

// One GRU over a 16-sample tile; writes per-step mean(h) into R[:, colBase+t].
static __device__ void gru16(const float* __restrict__ seqG, int rowBase,
                             const float* __restrict__ kern,   // (96,)
                             const float* __restrict__ bias,   // (2,96)
                             const v16h* __restrict__ Wg,      // 6 swizzled 32x16 tiles
                             float* seqL, _Float16* hbf, float* hf32,
                             _Float16* R, int colBase, int lane) {
  for (int idx = lane; idx < 512; idx += 32)
    seqL[idx] = seqG[(rowBase + (idx >> 5)) * 32 + (idx & 31)];
  for (int idx = lane; idx < 512; idx += 32) hbf[idx] = (_Float16)0.f;

  int nlo = lane & 15;
  int m0 = (lane >> 4) << 3;
  float k0[6], b0[6], b1[6];
#pragma unroll
  for (int q = 0; q < 6; ++q) {
    int j = q * 16 + nlo;           // rec output column; q: 0,1=z 2,3=r 4,5=h
    k0[q] = kern[j];
    b0[q] = bias[j];
    b1[q] = bias[96 + j];
  }
  float h0[8], h1[8];
#pragma unroll
  for (int r = 0; r < 8; ++r) { h0[r] = 0.f; h1[r] = 0.f; }

  WAVE_LDS_WAIT();
  for (int t = 0; t < 32; ++t) {
    v16h a = loadA(hbf, 32, 0, lane);                  // H as 16x32 f16
    v8f r0 = wmma32(a, Wg[0 * 32 + lane], zero8());
    v8f r1 = wmma32(a, Wg[1 * 32 + lane], zero8());
    v8f r2 = wmma32(a, Wg[2 * 32 + lane], zero8());
    v8f r3 = wmma32(a, Wg[3 * 32 + lane], zero8());
    v8f r4 = wmma32(a, Wg[4 * 32 + lane], zero8());
    v8f r5 = wmma32(a, Wg[5 * 32 + lane], zero8());
#pragma unroll
    for (int r = 0; r < 8; ++r) {
      int m = r + m0;
      float xs = seqL[m * 32 + t];
      // units nlo (tile 0): gates from rec tiles 0/2/4
      float z  = fast_sigm(xs * k0[0] + b0[0] + r0[r] + b1[0]);
      float rr = fast_sigm(xs * k0[2] + b0[2] + r2[r] + b1[2]);
      float hh = fast_tanh(xs * k0[4] + b0[4] + rr * (r4[r] + b1[4]));
      h0[r] = z * h0[r] + (1.f - z) * hh;
      // units 16+nlo (tile 1): gates from rec tiles 1/3/5
      z  = fast_sigm(xs * k0[1] + b0[1] + r1[r] + b1[1]);
      rr = fast_sigm(xs * k0[3] + b0[3] + r3[r] + b1[3]);
      hh = fast_tanh(xs * k0[5] + b0[5] + rr * (r5[r] + b1[5]));
      h1[r] = z * h1[r] + (1.f - z) * hh;

      hbf[m * 32 + nlo]       = (_Float16)h0[r];
      hbf[m * 32 + 16 + nlo]  = (_Float16)h1[r];
      hf32[m * 32 + nlo]      = h0[r];
      hf32[m * 32 + 16 + nlo] = h1[r];
    }
    WAVE_LDS_WAIT();
    if (lane < 16) {                                   // mean over 32 units
      float s = 0.f;
      const float4* hv = (const float4*)(hf32 + lane * 32);
#pragma unroll
      for (int q = 0; q < 8; ++q) { float4 v4 = hv[q]; s += v4.x + v4.y + v4.z + v4.w; }
      R[lane * 128 + colBase + t] = (_Float16)(s * (1.f / 32.f));
    }
  }
}

// ---------------------------------------------------------------------------
// Fused main kernel: one wave (32 threads) per 16-sample tile.
// ---------------------------------------------------------------------------
__global__ __launch_bounds__(32) void dlrm_main(Params p) {
  __shared__ WaveLDS lds[1];
  WaveLDS& w = lds[0];
  int lane = threadIdx.x & 31;
  int rowBase = blockIdx.x << 4;

  // ---- bottom MLP -------------------------------------------------------
  _Float16* denseH = w.hbf;                       // 16x32 f16 staging (K padded)
  for (int idx = lane; idx < 512; idx += 32) {
    int s = idx >> 5, d = idx & 31;
    float v = (d < 11) ? p.dense[(rowBase + s) * 11 + d] : 0.f;
    denseH[idx] = (_Float16)v;
  }
  WAVE_LDS_WAIT();
  gemm16(denseH, 32, 32, w.bufA, 512, 512, p.Wb0, p.b_bot0, p.sc_b0, p.sh_b0, true, lane);
  WAVE_LDS_WAIT();
  gemm16(w.bufA, 512, 512, w.bufB, 256, 256, p.Wb1, p.b_bot1, p.sc_b1, p.sh_b1, true, lane);
  WAVE_LDS_WAIT();
  gemm16(w.bufB, 256, 256, w.bufA, 64, 64, p.Wb2, p.b_bot2, p.sc_b2, p.sh_b2, true, lane);
  WAVE_LDS_WAIT();
  // last bottom layer: ReLU, no BN; write x into R cols 0..15
  gemm16(w.bufA, 64, 64, w.R, 128, 16, p.Wb3, p.b_bot3, nullptr, nullptr, true, lane);
  WAVE_LDS_WAIT();

  // ---- embeddings + pairwise interactions (VALU) ------------------------
  float* T = (float*)w.bufB;                      // 16 x 6 x 16 f32 (reuse bufB)
  for (int idx = lane; idx < 16 * 96; idx += 32) {
    int s = idx / 96, rem = idx - s * 96;
    int f = rem >> 4, d = rem & 15;
    float v;
    if (f < 5) {
      int ix = p.sparse[(rowBase + s) * 5 + f];
      v = p.emb[f][ix * 16 + d];
    } else {
      v = (float)w.R[s * 128 + d];                // x row
    }
    T[s * 96 + rem] = v;
  }
  WAVE_LDS_WAIT();
  const int iu[15] = {0,0,0,0,0,1,1,1,1,2,2,2,3,3,4};
  const int ju[15] = {1,2,3,4,5,2,3,4,5,3,4,5,4,5,5};
  for (int pi = lane; pi < 240; pi += 32) {
    int s = pi / 15, pr = pi - s * 15;
    const float* ti = T + s * 96 + iu[pr] * 16;
    const float* tj = T + s * 96 + ju[pr] * 16;
    float dot = 0.f;
#pragma unroll
    for (int d = 0; d < 16; ++d) dot += ti[d] * tj[d];
    w.R[s * 128 + 16 + pr] = (_Float16)dot;       // Z_flat -> cols 16..30
  }
  // duration/interact prefs + zero pad -> cols 95..110, 111..121, 122..127
  for (int idx = lane; idx < 256; idx += 32) {
    int s = idx >> 4, d = idx & 15;
    w.R[s * 128 + 95 + d] = (_Float16)p.dur[(rowBase + s) * 16 + d];
  }
  for (int idx = lane; idx < 176; idx += 32) {
    int s = idx / 11, d = idx - s * 11;
    w.R[s * 128 + 111 + d] = (_Float16)p.inter[(rowBase + s) * 11 + d];
  }
  for (int idx = lane; idx < 96; idx += 32) {
    int s = idx / 6, d = idx - s * 6;
    w.R[s * 128 + 122 + d] = (_Float16)0.f;
  }

  // ---- GRUs: lf -> cols 31..62, pf -> cols 63..94 ------------------------
  gru16(p.launch_seq, rowBase, p.gl_kern, p.gl_bias, p.Wgl,
        w.seq, w.hbf, w.hf32, w.R, 31, lane);
  gru16(p.play_seq, rowBase, p.gp_kern, p.gp_bias, p.Wgp,
        w.seq, w.hbf, w.hf32, w.R, 63, lane);
  WAVE_LDS_WAIT();

  // ---- top MLP -----------------------------------------------------------
  gemm16(w.R, 128, 128, w.bufA, 512, 512, p.Wt0, p.b_top0, p.sc_t0, p.sh_t0, true, lane);
  WAVE_LDS_WAIT();
  gemm16(w.bufA, 512, 512, w.bufB, 256, 256, p.Wt1, p.b_top1, p.sc_t1, p.sh_t1, true, lane);
  WAVE_LDS_WAIT();
  {
    v8f acc = zero8();
#pragma unroll
    for (int kt = 0; kt < 8; ++kt) {
      v16h a = loadA(w.bufB, 256, kt << 5, lane);
      v16h b = p.Wt2[kt * 32 + lane];
      acc = wmma32(a, b, acc);
    }
    float bs = p.b_top2[0];
    if ((lane & 15) == 0) {                       // only N=0 column valid
      int m0 = (lane >> 4) << 3;
#pragma unroll
      for (int r = 0; r < 8; ++r) p.out[rowBase + m0 + r] = acc[r] + bs;
    }
  }
}

// ---------------------------------------------------------------------------
// Host launcher
// ---------------------------------------------------------------------------
extern "C" void kernel_launch(void* const* d_in, const int* in_sizes, int n_in,
                              void* d_out, int out_size, void* d_ws, size_t ws_size,
                              hipStream_t stream) {
  auto f = [&](int i) { return (const float*)d_in[i]; };

  Params P;
  P.launch_seq = f(0);
  P.play_seq   = f(1);
  P.dur        = f(2);
  P.inter      = f(3);
  P.dense      = f(4);
  P.sparse     = (const int*)d_in[5];
  P.out        = (float*)d_out;

  const float *bW[4], *bb[4], *bg[3], *bbe[3], *bm[3], *bv[3];
  const float *tW[3], *tb[3], *tg[2], *tbe[2], *tm[2], *tv[2];
  const float *glk, *glr, *glb, *gpk, *gpr, *gpb;

  if (in_sizes[6] == 11 * 512) {
    // jax tree_leaves order: sorted dict keys: bot, emb, gru_launch, gru_play, top
    for (int l = 0; l < 3; ++l) {
      int base = 6 + l * 6;
      bW[l] = f(base); bb[l] = f(base + 1); bg[l] = f(base + 2);
      bbe[l] = f(base + 3); bm[l] = f(base + 4); bv[l] = f(base + 5);
    }
    bW[3] = f(24); bb[3] = f(25);
    for (int e = 0; e < 5; ++e) P.emb[e] = f(26 + e);
    glk = f(31); glr = f(32); glb = f(33);
    gpk = f(34); gpr = f(35); gpb = f(36);
  } else {
    // insertion order: emb, gru_launch, gru_play, bot, top
    for (int e = 0; e < 5; ++e) P.emb[e] = f(6 + e);
    glk = f(11); glr = f(12); glb = f(13);
    gpk = f(14); gpr = f(15); gpb = f(16);
    for (int l = 0; l < 3; ++l) {
      int base = 17 + l * 6;
      bW[l] = f(base); bb[l] = f(base + 1); bg[l] = f(base + 2);
      bbe[l] = f(base + 3); bm[l] = f(base + 4); bv[l] = f(base + 5);
    }
    bW[3] = f(35); bb[3] = f(36);
  }
  // top indices coincide in both orderings (start at 37)
  tW[0] = f(37); tb[0] = f(38); tg[0] = f(39); tbe[0] = f(40); tm[0] = f(41); tv[0] = f(42);
  tW[1] = f(43); tb[1] = f(44); tg[1] = f(45); tbe[1] = f(46); tm[1] = f(47); tv[1] = f(48);
  tW[2] = f(49); tb[2] = f(50);

  P.b_bot0 = bb[0]; P.b_bot1 = bb[1]; P.b_bot2 = bb[2]; P.b_bot3 = bb[3];
  P.b_top0 = tb[0]; P.b_top1 = tb[1]; P.b_top2 = tb[2];
  P.gl_kern = glk; P.gl_bias = glb; P.gp_kern = gpk; P.gp_bias = gpb;

  // ---- workspace carve-out ----------------------------------------------
  char* ws = (char*)d_ws;
  size_t off = 0;
  auto carve = [&](size_t bytes) {
    void* ptr = ws + off;
    off = (off + bytes + 255) & ~(size_t)255;
    return ptr;
  };
  _Float16* wb0 = (_Float16*)carve(1 * 32 * 1024);    // KT=1  NT=32 -> 32 tiles
  _Float16* wb1 = (_Float16*)carve(256 * 1024);       // 16x16 tiles
  _Float16* wb2 = (_Float16*)carve(32 * 1024);        // 8x4
  _Float16* wb3 = (_Float16*)carve(2 * 1024);         // 2x1
  _Float16* wt0 = (_Float16*)carve(128 * 1024);       // 4x32
  _Float16* wt1 = (_Float16*)carve(256 * 1024);       // 16x16
  _Float16* wt2 = (_Float16*)carve(8 * 1024);         // 8x1
  _Float16* wgl = (_Float16*)carve(6 * 1024);         // 1x6
  _Float16* wgp = (_Float16*)carve(6 * 1024);         // 1x6
  float* sc_b0 = (float*)carve(512 * 4); float* sh_b0 = (float*)carve(512 * 4);
  float* sc_b1 = (float*)carve(256 * 4); float* sh_b1 = (float*)carve(256 * 4);
  float* sc_b2 = (float*)carve(64 * 4);  float* sh_b2 = (float*)carve(64 * 4);
  float* sc_t0 = (float*)carve(512 * 4); float* sh_t0 = (float*)carve(512 * 4);
  float* sc_t1 = (float*)carve(256 * 4); float* sh_t1 = (float*)carve(256 * 4);

  P.Wb0 = (const v16h*)wb0; P.Wb1 = (const v16h*)wb1;
  P.Wb2 = (const v16h*)wb2; P.Wb3 = (const v16h*)wb3;
  P.Wt0 = (const v16h*)wt0; P.Wt1 = (const v16h*)wt1; P.Wt2 = (const v16h*)wt2;
  P.Wgl = (const v16h*)wgl; P.Wgp = (const v16h*)wgp;
  P.sc_b0 = sc_b0; P.sh_b0 = sh_b0; P.sc_b1 = sc_b1; P.sh_b1 = sh_b1;
  P.sc_b2 = sc_b2; P.sh_b2 = sh_b2;
  P.sc_t0 = sc_t0; P.sh_t0 = sh_t0; P.sc_t1 = sc_t1; P.sh_t1 = sh_t1;

  auto gblk = [](int total) { int g = (total + 255) / 256; return g > 1024 ? 1024 : g; };

  // ---- weight prep -------------------------------------------------------
  swizzle_w<<<gblk(32 * 512), 256, 0, stream>>>(bW[0], wb0, 11, 512, 1, 32);
  swizzle_w<<<gblk(256 * 512), 256, 0, stream>>>(bW[1], wb1, 512, 256, 16, 16);
  swizzle_w<<<gblk(32 * 512), 256, 0, stream>>>(bW[2], wb2, 256, 64, 8, 4);
  swizzle_w<<<gblk(2 * 512), 256, 0, stream>>>(bW[3], wb3, 64, 16, 2, 1);
  swizzle_w<<<gblk(128 * 512), 256, 0, stream>>>(tW[0], wt0, 122, 512, 4, 32);
  swizzle_w<<<gblk(256 * 512), 256, 0, stream>>>(tW[1], wt1, 512, 256, 16, 16);
  swizzle_w<<<gblk(8 * 512), 256, 0, stream>>>(tW[2], wt2, 256, 1, 8, 1);
  swizzle_w<<<gblk(6 * 512), 256, 0, stream>>>(glr, wgl, 32, 96, 1, 6);
  swizzle_w<<<gblk(6 * 512), 256, 0, stream>>>(gpr, wgp, 32, 96, 1, 6);

  bn_prep<<<2, 256, 0, stream>>>(bg[0], bbe[0], bm[0], bv[0], sc_b0, sh_b0, 512);
  bn_prep<<<1, 256, 0, stream>>>(bg[1], bbe[1], bm[1], bv[1], sc_b1, sh_b1, 256);
  bn_prep<<<1, 256, 0, stream>>>(bg[2], bbe[2], bm[2], bv[2], sc_b2, sh_b2, 64);
  bn_prep<<<2, 256, 0, stream>>>(tg[0], tbe[0], tm[0], tv[0], sc_t0, sh_t0, 512);
  bn_prep<<<1, 256, 0, stream>>>(tg[1], tbe[1], tm[1], tv[1], sc_t1, sh_t1, 256);

  // ---- fused model: 65536 samples / 16 per wave = 4096 blocks ------------
  dlrm_main<<<4096, 32, 0, stream>>>(P);

  (void)n_in; (void)out_size; (void)ws_size; (void)in_sizes;
}
